// ResBlock_3d_2_1786706395363
// MI455X (gfx1250) — compile-verified, matrix operand order checked by reading, not run on hardware
//
#include <hip/hip_runtime.h>
#include <hip/hip_bf16.h>
#include <math.h>

typedef _Float16 half_t;
typedef __attribute__((ext_vector_type(8)))  _Float16 v8h;
typedef __attribute__((ext_vector_type(16))) _Float16 v16h;
typedef __attribute__((ext_vector_type(8)))  float    v8f;

#define C_   64
#define H_   64
#define W_   64
#define T_   16
#define HWT_ 65536   /* H*W*T */
#define KD_  576     /* 9*C   */

__device__ __forceinline__ float mishf(float x) {
    float sp = (x > 20.f) ? x : log1pf(expf(x));
    return x * tanhf(sp);
}

// ---------------------------------------------------------------------------
// Pack fp32 OIHW(9) weights into f16 [N][K] (K = tap*64 + cin), offset conv
// padded from 18 to 32 output rows so both GEMMs tile as N multiples of 16.
// ---------------------------------------------------------------------------
__global__ void prep_weights(const float* __restrict__ w0,  const float* __restrict__ w1,
                             const float* __restrict__ wo0, const float* __restrict__ wo1,
                             half_t* __restrict__ wT0, half_t* __restrict__ wT1,
                             half_t* __restrict__ wO0, half_t* __restrict__ wO1) {
    const int WMAIN = 64 * KD_;   // 36864
    const int WOFF  = 32 * KD_;   // 18432
    int idx = blockIdx.x * blockDim.x + threadIdx.x;
    if (idx < WMAIN) {
        int o = idx / KD_, K = idx % KD_, k = K >> 6, i = K & 63;
        wT0[idx] = (half_t)w0[(o * 64 + i) * 9 + k];
    } else if (idx < 2 * WMAIN) {
        int r = idx - WMAIN;
        int o = r / KD_, K = r % KD_, k = K >> 6, i = K & 63;
        wT1[r] = (half_t)w1[(o * 64 + i) * 9 + k];
    } else if (idx < 2 * WMAIN + WOFF) {
        int r = idx - 2 * WMAIN;
        int o = r / KD_, K = r % KD_, k = K >> 6, i = K & 63;
        wO0[r] = (o < 18) ? (half_t)wo0[(o * 64 + i) * 9 + k] : (half_t)0.f;
    } else if (idx < 2 * WMAIN + 2 * WOFF) {
        int r = idx - 2 * WMAIN - WOFF;
        int o = r / KD_, K = r % KD_, k = K >> 6, i = K & 63;
        wO1[r] = (o < 18) ? (half_t)wo1[(o * 64 + i) * 9 + k] : (half_t)0.f;
    }
}

// NCHWD -> position-major [HWT][C] fp32 (GEMM-friendly; channel contiguous)
__global__ void transpose_in(const float* __restrict__ x, float* __restrict__ xT) {
    int idx = blockIdx.x * blockDim.x + threadIdx.x;   // HWT*64 total
    if (idx >= HWT_ * 64) return;
    int m = idx >> 6, c = idx & 63;
    xT[idx] = x[c * HWT_ + m];
}

// ---------------------------------------------------------------------------
// Offset conv: off[m][0..17] = bO + sum_K vbuf[m][K] * wO[n][K]
// Block = 128 threads (4 waves) = 2 Msub x 2 Nsub tiles of 16x16. Mtile = 32.
// ---------------------------------------------------------------------------
__global__ void offset_conv(const float* __restrict__ act,
                            const half_t* __restrict__ wO,
                            const float* __restrict__ bO,
                            float* __restrict__ off) {
    __shared__ half_t vbuf[32][KD_];
    const int tid = threadIdx.x;
    const int mBase = blockIdx.x * 32;
    __builtin_prefetch(wO, 0, 0);

    // phase 1: im2col of regular 3x3 neighborhood (zero padded in H/W)
    for (int pk = tid; pk < 32 * 9; pk += blockDim.x) {
        int p = pk / 9, k = pk % 9;
        int m = mBase + p;
        int h = m >> 10, w = (m >> 4) & 63, t = m & 15;
        int hh = h + (k / 3) - 1, ww = w + (k % 3) - 1;
        half_t* dst = &vbuf[p][k << 6];
        if (hh >= 0 && hh < H_ && ww >= 0 && ww < W_) {
            const float* src = act + ((((hh << 6) + ww) << 4) + t) * 64;
            for (int c = 0; c < 64; c += 4) {
                float4 f = *(const float4*)(src + c);
                dst[c + 0] = (half_t)f.x; dst[c + 1] = (half_t)f.y;
                dst[c + 2] = (half_t)f.z; dst[c + 3] = (half_t)f.w;
            }
        } else {
            for (int c = 0; c < 64; ++c) dst[c] = (half_t)0.f;
        }
    }
    __syncthreads();

    // phase 2: WMMA over K=576 (18 chunks of 32)
    const int lane = tid & 31, wave = tid >> 5;
    const int msub = wave >> 1, nsub = wave & 1;
    const int hi8  = (lane >> 4) << 3;                 // 0 | 8: ISA K-group split
    const half_t* ap = &vbuf[msub * 16 + (lane & 15)][0];
    const half_t* bp = wO + (nsub * 16 + (lane & 15)) * KD_;

    v8f acc = {};
    #pragma unroll
    for (int kb = 0; kb < KD_; kb += 32) {
        v16h a, b;
        ((v8h*)&a)[0] = *(const v8h*)(ap + kb + hi8);
        ((v8h*)&a)[1] = *(const v8h*)(ap + kb + 16 + hi8);
        ((v8h*)&b)[0] = *(const v8h*)(bp + kb + hi8);
        ((v8h*)&b)[1] = *(const v8h*)(bp + kb + 16 + hi8);
        acc = __builtin_amdgcn_wmma_f32_16x16x32_f16(false, a, false, b,
                                                     (short)0, acc, false, false);
    }

    const int n = nsub * 16 + (lane & 15);
    if (n < 18) {
        float bias = bO[n];
        #pragma unroll
        for (int r = 0; r < 8; ++r) {
            int M = msub * 16 + r + ((lane >> 4) << 3);  // C/D layout: M=r(+8), N=lane%16
            off[(mBase + M) * 18 + n] = acc[r] + bias;
        }
    }
}

// ---------------------------------------------------------------------------
// Deformable conv: bilinear-gathered im2col into LDS, then WMMA.
// Block = 256 threads (8 waves) = 2 Msub x 4 Nsub. Mtile = 32, N = 64.
// mode 0: mish -> outPM (position-major). mode 1: +bias +residual -> NCHWD.
// ---------------------------------------------------------------------------
__global__ void deform_conv(const float* __restrict__ act,
                            const float* __restrict__ off,
                            const half_t* __restrict__ wT,
                            const float* __restrict__ bias,
                            const float* __restrict__ resid,
                            float* __restrict__ outPM,
                            float* __restrict__ outFinal,
                            int mode) {
    __shared__ half_t vbuf[32][KD_];
    const int tid = threadIdx.x;
    const int mBase = blockIdx.x * 32;
    __builtin_prefetch(wT, 0, 0);

    // phase 1: bilinear sample 9 taps x 64 channels per position
    for (int pk = tid; pk < 32 * 9; pk += blockDim.x) {
        int p = pk / 9, k = pk % 9;
        int m = mBase + p;
        int h = m >> 10, w = (m >> 4) & 63, t = m & 15;
        float ph = (float)(h + (k / 3) - 1) + off[m * 18 + k];
        float pw = (float)(w + (k % 3) - 1) + off[m * 18 + 9 + k];
        float h0f = floorf(ph), w0f = floorf(pw);
        float lh = ph - h0f, lw = pw - w0f;
        int h0 = (int)h0f, w0 = (int)w0f;
        float c00 = (1.f - lh) * (1.f - lw), c01 = (1.f - lh) * lw;
        float c10 = lh * (1.f - lw),         c11 = lh * lw;
        bool vh0 = (h0 >= 0) && (h0 < H_);
        bool vh1 = (h0 + 1 >= 0) && (h0 + 1 < H_);
        bool vw0 = (w0 >= 0) && (w0 < W_);
        bool vw1 = (w0 + 1 >= 0) && (w0 + 1 < W_);
        const float* p00 = (vh0 && vw0) ? act + (((((h0    ) << 6) + w0    ) << 4) + t) * 64 : (const float*)0;
        const float* p01 = (vh0 && vw1) ? act + (((((h0    ) << 6) + w0 + 1) << 4) + t) * 64 : (const float*)0;
        const float* p10 = (vh1 && vw0) ? act + (((((h0 + 1) << 6) + w0    ) << 4) + t) * 64 : (const float*)0;
        const float* p11 = (vh1 && vw1) ? act + (((((h0 + 1) << 6) + w0 + 1) << 4) + t) * 64 : (const float*)0;
        half_t* dst = &vbuf[p][k << 6];
        for (int c = 0; c < 64; c += 4) {
            float r0 = 0.f, r1 = 0.f, r2 = 0.f, r3 = 0.f;
            if (p00) { float4 f = *(const float4*)(p00 + c); r0 += c00 * f.x; r1 += c00 * f.y; r2 += c00 * f.z; r3 += c00 * f.w; }
            if (p01) { float4 f = *(const float4*)(p01 + c); r0 += c01 * f.x; r1 += c01 * f.y; r2 += c01 * f.z; r3 += c01 * f.w; }
            if (p10) { float4 f = *(const float4*)(p10 + c); r0 += c10 * f.x; r1 += c10 * f.y; r2 += c10 * f.z; r3 += c10 * f.w; }
            if (p11) { float4 f = *(const float4*)(p11 + c); r0 += c11 * f.x; r1 += c11 * f.y; r2 += c11 * f.z; r3 += c11 * f.w; }
            dst[c + 0] = (half_t)r0; dst[c + 1] = (half_t)r1;
            dst[c + 2] = (half_t)r2; dst[c + 3] = (half_t)r3;
        }
    }
    __syncthreads();

    // phase 2: WMMA over K=576
    const int lane = tid & 31, wave = tid >> 5;
    const int msub = wave >> 2, nsub = wave & 3;
    const int hi8  = (lane >> 4) << 3;
    const half_t* ap = &vbuf[msub * 16 + (lane & 15)][0];
    const half_t* bp = wT + (nsub * 16 + (lane & 15)) * KD_;

    v8f acc = {};
    #pragma unroll
    for (int kb = 0; kb < KD_; kb += 32) {
        v16h a, b;
        ((v8h*)&a)[0] = *(const v8h*)(ap + kb + hi8);
        ((v8h*)&a)[1] = *(const v8h*)(ap + kb + 16 + hi8);
        ((v8h*)&b)[0] = *(const v8h*)(bp + kb + hi8);
        ((v8h*)&b)[1] = *(const v8h*)(bp + kb + 16 + hi8);
        acc = __builtin_amdgcn_wmma_f32_16x16x32_f16(false, a, false, b,
                                                     (short)0, acc, false, false);
    }

    const int n = nsub * 16 + (lane & 15);
    const float bb = bias[n];
    #pragma unroll
    for (int r = 0; r < 8; ++r) {
        int m = mBase + msub * 16 + r + ((lane >> 4) << 3);
        float val = acc[r] + bb;
        if (mode == 0) {
            outPM[m * 64 + n] = mishf(val);
        } else {
            outFinal[n * HWT_ + m] = val + resid[n * HWT_ + m];
        }
    }
}

// ---------------------------------------------------------------------------
extern "C" void kernel_launch(void* const* d_in, const int* in_sizes, int n_in,
                              void* d_out, int out_size, void* d_ws, size_t ws_size,
                              hipStream_t stream) {
    (void)in_sizes; (void)n_in; (void)out_size; (void)ws_size;
    const float* x      = (const float*)d_in[0];
    const float* w_off0 = (const float*)d_in[1];
    const float* b_off0 = (const float*)d_in[2];
    const float* w0     = (const float*)d_in[3];
    const float* b0     = (const float*)d_in[4];
    const float* w_off1 = (const float*)d_in[5];
    const float* b_off1 = (const float*)d_in[6];
    const float* w1     = (const float*)d_in[7];
    const float* b1     = (const float*)d_in[8];
    float* out = (float*)d_out;

    char* ws = (char*)d_ws;
    const size_t SZ_ACT = (size_t)HWT_ * 64 * sizeof(float);   // 16 MB
    const size_t SZ_OFF = (size_t)HWT_ * 18 * sizeof(float);   // 4.7 MB
    float*  xT  = (float*)(ws);
    float*  hP  = (float*)(ws + SZ_ACT);
    float*  off = (float*)(ws + 2 * SZ_ACT);
    half_t* wT0 = (half_t*)(ws + 2 * SZ_ACT + SZ_OFF);
    half_t* wT1 = wT0 + 64 * KD_;
    half_t* wO0 = wT1 + 64 * KD_;
    half_t* wO1 = wO0 + 32 * KD_;

    prep_weights<<<(2 * 64 * KD_ + 2 * 32 * KD_ + 255) / 256, 256, 0, stream>>>(
        w0, w1, w_off0, w_off1, wT0, wT1, wO0, wO1);
    transpose_in<<<(HWT_ * 64) / 256, 256, 0, stream>>>(x, xT);

    // layer 0
    offset_conv<<<HWT_ / 32, 128, 0, stream>>>(xT, wO0, b_off0, off);
    deform_conv<<<HWT_ / 32, 256, 0, stream>>>(xT, off, wT0, b0,
                                               (const float*)0, hP, (float*)0, 0);
    // layer 1 + residual
    offset_conv<<<HWT_ / 32, 128, 0, stream>>>(hP, wO1, b_off1, off);
    deform_conv<<<HWT_ / 32, 256, 0, stream>>>(hP, off, wT1, b1,
                                               x, (float*)0, out, 1);
}
